// AdaptiveInteraction_2207613190274
// MI455X (gfx1250) — matrix operand order repaired and checked by reading, not compiled, not run on hardware
//
#include <hip/hip_runtime.h>
#include <hip/hip_bf16.h>
#include <math.h>

typedef __attribute__((ext_vector_type(16))) _Float16 v16h;
typedef __attribute__((ext_vector_type(8)))  _Float16 v8h;
typedef __attribute__((ext_vector_type(8)))  float    v8f;

#define B_    2
#define N_    3000
#define NPAD_ 3008          // 188 chunks of 16
#define D_    64
#define K_    2
#define LDSW_ 3016          // padded row stride in floats (bank-spread: 3016%64==8; 3016%4==0)
#define NCHUNK_ (NPAD_/16)  // 188

// ---------------------------------------------------------------- K1: Ein = input @ W^T + b  (f32 + f16 copy, zero-padded rows)
__global__ void k_ein(const float* __restrict__ in, const float* __restrict__ W,
                      const float* __restrict__ bias, float* __restrict__ Ein,
                      _Float16* __restrict__ Einh) {
  __shared__ float row[4 * 64];
  int b = blockIdx.y;
  int t = threadIdx.x;
  int d = t & 63, rl = t >> 6;
  int n = blockIdx.x * 4 + rl;
  bool valid = (n < N_);
  if (valid) row[rl * 64 + d] = in[((size_t)b * N_ + n) * 64 + d];
  __syncthreads();
  if (valid) {
    float acc = bias[d];
#pragma unroll
    for (int c = 0; c < 64; ++c) acc = fmaf(row[rl * 64 + c], W[d * 64 + c], acc);
    Ein[((size_t)b * N_ + n) * 64 + d] = acc;
    Einh[((size_t)b * NPAD_ + n) * 64 + d] = (_Float16)acc;
  } else if (n < NPAD_) {
    Einh[((size_t)b * NPAD_ + n) * 64 + d] = (_Float16)0.f;
  }
}

// ---------------------------------------------------------------- K2: G[b] = Ein^T Ein / sqrt(64)   (64x64 per batch, atomic accumulate)
#define GRAM_SPLIT 25
#define GRAM_ROWS  120      // 3000/25
__global__ void k_gram(const float* __restrict__ Ein, float* __restrict__ G) {
  __shared__ float tile[16 * 64];
  int b = blockIdx.y;
  int n_begin = blockIdx.x * GRAM_ROWS;
  int n_end = min(N_, n_begin + GRAM_ROWS);
  int t = threadIdx.x;
  int c1 = t >> 2;
  int c2b = (t & 3) * 16;
  float acc[16];
#pragma unroll
  for (int q = 0; q < 16; ++q) acc[q] = 0.f;
  for (int nb = n_begin; nb < n_end; nb += 16) {
    int cnt = min(16, n_end - nb);
    for (int idx = t; idx < cnt * 64; idx += 256)
      tile[idx] = Ein[((size_t)b * N_ + nb) * 64 + idx];
    __syncthreads();
    for (int r = 0; r < cnt; ++r) {
      float e1 = tile[r * 64 + c1];
#pragma unroll
      for (int q = 0; q < 16; ++q) acc[q] = fmaf(e1, tile[r * 64 + c2b + q], acc[q]);
    }
    __syncthreads();
  }
#pragma unroll
  for (int q = 0; q < 16; ++q)
    atomicAdd(&G[((size_t)b * 64 + c1) * 64 + c2b + q], acc[q] * 0.125f);
}

// ---------------------------------------------------------------- K3: BN stats of E = Ein*G (E never stored)
__global__ void k_stats(const float* __restrict__ Ein, const float* __restrict__ G,
                        float* __restrict__ stats) {
  __shared__ float Gs[64 * 64];
  __shared__ float er[4 * 64];
  __shared__ float ssum[64], ssq[64];
  int b = blockIdx.y, t = threadIdx.x;
  int d = t & 63, rl = t >> 6;
  int n = blockIdx.x * 4 + rl;
  for (int i = t; i < 4096; i += 256) Gs[i] = G[(size_t)b * 4096 + i];
  if (t < 64) { ssum[t] = 0.f; ssq[t] = 0.f; }
  if (n < N_) er[rl * 64 + d] = Ein[((size_t)b * N_ + n) * 64 + d];
  __syncthreads();
  if (n < N_) {
    float e = 0.f;
#pragma unroll
    for (int c = 0; c < 64; ++c) e = fmaf(er[rl * 64 + c], Gs[c * 64 + d], e);
    atomicAdd(&ssum[d], e);
    atomicAdd(&ssq[d], e * e);
  }
  __syncthreads();
  if (t < 64) { atomicAdd(&stats[t], ssum[t]); atomicAdd(&stats[64 + t], ssq[t]); }
}

// ---------------------------------------------------------------- K4: s/shift, M[b] = G diag(s) G^T, gv[b] = G*shift  (one tiny block)
__global__ void k_final(const float* __restrict__ G, const float* __restrict__ stats,
                        const float* __restrict__ gamma, const float* __restrict__ beta,
                        float* __restrict__ M, float* __restrict__ gv) {
  __shared__ float s_[64], sh_[64], Gsh[4096];
  int t = threadIdx.x;  // 64 threads
  float invn = 1.0f / (float)(B_ * N_);
  float mean = stats[t] * invn;
  float var = stats[64 + t] * invn - mean * mean;
  float s = gamma[t] * rsqrtf(var + 1e-5f);
  float shift = beta[t] - mean * s;
  s_[t] = s; sh_[t] = shift;
  __syncthreads();
  for (int b = 0; b < B_; ++b) {
    for (int i = t; i < 4096; i += 64) Gsh[i] = G[(size_t)b * 4096 + i];
    __syncthreads();
    float mrow[64];
#pragma unroll
    for (int j = 0; j < 64; ++j) mrow[j] = 0.f;
    float gvacc = 0.f;
    for (int c = 0; c < 64; ++c) {
      float gic = Gsh[t * 64 + c];
      float wq = gic * s_[c];
      gvacc = fmaf(gic, sh_[c], gvacc);
#pragma unroll
      for (int j = 0; j < 64; ++j) mrow[j] = fmaf(wq, Gsh[j * 64 + c], mrow[j]);
    }
#pragma unroll
    for (int j = 0; j < 64; ++j) M[(size_t)b * 4096 + t * 64 + j] = mrow[j];
    gv[b * 64 + t] = gvacc;
    __syncthreads();
  }
}

// ---------------------------------------------------------------- K5: P = Ein*M (f16, padded), w = Ein*(G*shift) (padded)
__global__ void k_pw(const float* __restrict__ Ein, const float* __restrict__ M,
                     const float* __restrict__ gv, _Float16* __restrict__ Ph,
                     float* __restrict__ w) {
  __shared__ float Ms[4096];
  __shared__ float er[4 * 64];
  __shared__ float gvs[64];
  int b = blockIdx.y, t = threadIdx.x;
  int d = t & 63, rl = t >> 6;
  int n = blockIdx.x * 4 + rl;
  for (int i = t; i < 4096; i += 256) Ms[i] = M[(size_t)b * 4096 + i];
  if (t < 64) gvs[t] = gv[b * 64 + t];
  if (n < N_) er[rl * 64 + d] = Ein[((size_t)b * N_ + n) * 64 + d];
  __syncthreads();
  if (n < N_) {
    float p = 0.f;
#pragma unroll
    for (int c = 0; c < 64; ++c) p = fmaf(er[rl * 64 + c], Ms[c * 64 + d], p);
    Ph[((size_t)b * NPAD_ + n) * 64 + d] = (_Float16)p;
    if (d == 0) {
      float wa = 0.f;
#pragma unroll
      for (int c = 0; c < 64; ++c) wa = fmaf(er[rl * 64 + c], gvs[c], wa);
      w[(size_t)b * NPAD_ + n] = wa;
    }
  } else if (n < NPAD_) {
    Ph[((size_t)b * NPAD_ + n) * 64 + d] = (_Float16)0.f;
    if (d == 0) w[(size_t)b * NPAD_ + n] = 0.f;
  }
}

// ---------------------------------------------------------------- K6: WMMA  A = P*Ein^T + w, relu, softmax, out[k,b,i,j] = m[k,j]*A
// A-operand f16 16x32 layout (ISA 7.12.2): lane<16 -> M=lane, K in [h*8,h*8+8) and +16
__device__ __forceinline__ v16h load_a16(const _Float16* __restrict__ p, int koff) {
  v8h x = *(const v8h*)(p + koff);
  v8h y = *(const v8h*)(p + koff + 16);
  v16h r;
#pragma unroll
  for (int i = 0; i < 8; ++i) { r[i] = x[i]; r[i + 8] = y[i]; }
  return r;
}

__global__ void __launch_bounds__(128) k_attn(const _Float16* __restrict__ Ph,
                                              const _Float16* __restrict__ Einh,
                                              const float* __restrict__ w,
                                              const float* __restrict__ m,
                                              float* __restrict__ out) {
  extern __shared__ float smem[];
  float* alds   = smem;                  // 16 * LDSW_
  float* redbuf = smem + 16 * LDSW_;     // 128
  float* rowmax = redbuf + 128;          // 16
  float* rowinv = rowmax + 16;           // 16

  int b = blockIdx.y;
  int i0 = blockIdx.x * 16;
  int t = threadIdx.x;
  int wave = t >> 5;
  int lane = t & 31;
  int half = lane >> 4;
  int mrow = lane & 15;

  // Per-wave A operand: 16x64 P tile, split into two K=32 operands.
  const _Float16* prow = Ph + ((size_t)b * NPAD_ + (i0 + mrow)) * 64;
  v16h a_lo = load_a16(prow,      half * 8);
  v16h a_hi = load_a16(prow + 32, half * 8);

  // Stream j-chunks of 16 columns; 4 waves stride the chunk space, 2 tiles/iter.
#pragma unroll 2
  for (int ch = wave; ch < NCHUNK_; ch += 4) {
    int j = ch * 16 + mrow;
    const _Float16* brow = Einh + ((size_t)b * NPAD_ + j) * 64;
    __builtin_prefetch((const void*)(brow + 64 * 64), 0, 3);  // next chunk for this wave, near-cache
    // B operand f16 32x16, same K order as A: lane<16 -> K=0..7,16..23 col=lane; lane>=16 -> K=8..15,24..31
    v16h b_lo = load_a16(brow,      half * 8);
    v16h b_hi = load_a16(brow + 32, half * 8);
    float wv = w[(size_t)b * NPAD_ + j];
    v8f c = {};
    c = __builtin_amdgcn_wmma_f32_16x16x32_f16(false, a_lo, false, b_lo, (short)0, c, false, false);
    c = __builtin_amdgcn_wmma_f32_16x16x32_f16(false, a_hi, false, b_hi, (short)0, c, false, false);
    // C/D layout: VGPR r -> M = r + half*8, N = lane&15
#pragma unroll
    for (int r = 0; r < 8; ++r) {
      float v = fmaxf(c[r] + wv, 0.f);   // + broadcast w[j], relu
      alds[(r + half * 8) * LDSW_ + j] = v;
    }
  }
  __syncthreads();

  // softmax over j (valid cols only): rowwise max
  int row = t >> 3, lr = t & 7;
  float lmax = 0.f;
  for (int j = lr; j < N_; j += 8) lmax = fmaxf(lmax, alds[row * LDSW_ + j]);
  redbuf[t] = lmax;
  __syncthreads();
  if (t < 16) {
    float mx = redbuf[t * 8];
#pragma unroll
    for (int q = 1; q < 8; ++q) mx = fmaxf(mx, redbuf[t * 8 + q]);
    rowmax[t] = mx;
  }
  __syncthreads();

  // exp and sum (exp stored back in LDS)
  float rm = rowmax[row];
  float lsum = 0.f;
  for (int j = lr; j < N_; j += 8) {
    float e = __expf(alds[row * LDSW_ + j] - rm);
    alds[row * LDSW_ + j] = e;
    lsum += e;
  }
  redbuf[t] = lsum;
  __syncthreads();
  if (t < 16) {
    float sm = 0.f;
#pragma unroll
    for (int q = 0; q < 8; ++q) sm += redbuf[t * 8 + q];
    rowinv[t] = 1.0f / sm;
  }
  __syncthreads();

  // write both k planes, float4-wide and coalesced along j (N_ % 4 == 0)
  const float4* m0 = (const float4*)m;
  const float4* m1 = (const float4*)(m + N_);
  for (int r = 0; r < 16; ++r) {
    int i = i0 + r;
    if (i >= N_) break;
    float inv = rowinv[r];
    const float4* arow = (const float4*)(alds + r * LDSW_);
    float4* o0 = (float4*)(out + (((size_t)0 * B_ + b) * N_ + i) * (size_t)N_);
    float4* o1 = (float4*)(out + (((size_t)1 * B_ + b) * N_ + i) * (size_t)N_);
    for (int q = t; q < N_ / 4; q += 128) {
      float4 a4 = arow[q];
      float4 w0 = m0[q], w1 = m1[q];
      float4 r0, r1;
      r0.x = w0.x * (a4.x * inv); r0.y = w0.y * (a4.y * inv);
      r0.z = w0.z * (a4.z * inv); r0.w = w0.w * (a4.w * inv);
      r1.x = w1.x * (a4.x * inv); r1.y = w1.y * (a4.y * inv);
      r1.z = w1.z * (a4.z * inv); r1.w = w1.w * (a4.w * inv);
      o0[q] = r0;
      o1[q] = r1;
    }
  }
}

// ---------------------------------------------------------------- host
static inline size_t align256(size_t x) { return (x + 255) & ~(size_t)255; }

extern "C" void kernel_launch(void* const* d_in, const int* in_sizes, int n_in,
                              void* d_out, int out_size, void* d_ws, size_t ws_size,
                              hipStream_t stream) {
  (void)in_sizes; (void)n_in; (void)out_size; (void)ws_size;
  const float* in    = (const float*)d_in[0];
  const float* m     = (const float*)d_in[1];
  const float* W     = (const float*)d_in[2];
  const float* bias  = (const float*)d_in[3];
  const float* gamma = (const float*)d_in[4];
  const float* beta  = (const float*)d_in[5];
  float* out = (float*)d_out;

  char* ws = (char*)d_ws;
  size_t off = 0;
  float*    Ein   = (float*)(ws + off);   off = align256(off + (size_t)B_ * N_ * 64 * 4);
  _Float16* Einh  = (_Float16*)(ws + off); off = align256(off + (size_t)B_ * NPAD_ * 64 * 2);
  float*    G     = (float*)(ws + off);   off = align256(off + (size_t)B_ * 4096 * 4);
  float*    stats = (float*)(ws + off);   off = align256(off + 128 * 4);
  float*    M     = (float*)(ws + off);   off = align256(off + (size_t)B_ * 4096 * 4);
  float*    gv    = (float*)(ws + off);   off = align256(off + (size_t)B_ * 64 * 4);
  _Float16* Ph    = (_Float16*)(ws + off); off = align256(off + (size_t)B_ * NPAD_ * 64 * 2);
  float*    wvec  = (float*)(ws + off);   off = align256(off + (size_t)B_ * NPAD_ * 4);

  hipMemsetAsync(G, 0, (size_t)B_ * 4096 * 4, stream);
  hipMemsetAsync(stats, 0, 128 * 4, stream);

  k_ein  <<<dim3(NPAD_ / 4, B_), 256, 0, stream>>>(in, W, bias, Ein, Einh);
  k_gram <<<dim3(GRAM_SPLIT, B_), 256, 0, stream>>>(Ein, G);
  k_stats<<<dim3((N_ + 3) / 4, B_), 256, 0, stream>>>(Ein, G, stats);
  k_final<<<1, 64, 0, stream>>>(G, stats, gamma, beta, M, gv);
  k_pw   <<<dim3(NPAD_ / 4, B_), 256, 0, stream>>>(Ein, M, gv, Ph, wvec);

  size_t smem = (size_t)(16 * LDSW_ + 128 + 16 + 16) * sizeof(float); // ~193.7 KB
  k_attn <<<dim3(NCHUNK_, B_), 128, smem, stream>>>(Ph, Einh, wvec, m, out);
}
